// Decoder_85289460564700
// MI455X (gfx1250) — compile-verified, matrix operand order checked by reading, not run on hardware
//
#include <hip/hip_runtime.h>
#include <hip/hip_bf16.h>
#include <stdint.h>

// ---------------------------------------------------------------------------
// Tacotron-style decoder for MI455X (gfx1250).
//   - batch tiles of M=16 -> 8 independent workgroups, each runs the full
//     256-step recurrence (recurrence is per batch row: no cross-WG sync).
//   - all GEMMs via v_wmma_f32_16x16x32_bf16 (bf16 in, f32 accum).
//   - keys (f32) + bf16 encoder copy precomputed once; both L2-resident.
// ---------------------------------------------------------------------------

typedef __attribute__((ext_vector_type(16))) __bf16 bf16x16;
typedef __attribute__((ext_vector_type(8)))  float  f32x8;

#define B_TOT 128
#define TDEC  256
#define TENC  512
#define HID   256
#define INR   400
#define INRP  416

static constexpr size_t KEYS_BYTES  = (size_t)B_TOT * TENC * HID * 4;  // 64 MB
static constexpr size_t ENCBF_BYTES = (size_t)B_TOT * TENC * HID * 2;  // 32 MB
static constexpr size_t ALIGN_OFF   = (size_t)B_TOT * TDEC * INR;      // out elems

// bf16 weight arena offsets (elements)
static constexpr size_t OFF_W1   = 0;                       // 256 x 416 (padded)
static constexpr size_t OFF_W2   = OFF_W1   + 256 * 416;    // 128 x 256
static constexpr size_t OFF_AIH  = OFF_W2   + 128 * 256;    // 768 x 384
static constexpr size_t OFF_AHH  = OFF_AIH  + 768 * 384;    // 768 x 256
static constexpr size_t OFF_WQ   = OFF_AHH  + 768 * 256;    // 256 x 256
static constexpr size_t OFF_WK   = OFF_WQ   + 256 * 256;    // 256 x 256
static constexpr size_t OFF_WOUT = OFF_WK   + 256 * 256;    // 256 x 512
static constexpr size_t OFF_D1IH = OFF_WOUT + 256 * 512;    // 768 x 256
static constexpr size_t OFF_D1HH = OFF_D1IH + 768 * 256;
static constexpr size_t OFF_D2IH = OFF_D1HH + 768 * 256;
static constexpr size_t OFF_D2HH = OFF_D2IH + 768 * 256;
static constexpr size_t OFF_PROJ = OFF_D2HH + 768 * 256;    // 400 x 256

union FragU { bf16x16 v; uint4 q[2]; };
union Bf8U  { uint4 q; __bf16 h[8]; };

// ISA 7.12.2 16-bit fragment mapping: element e of the per-lane v16bf holds
// k = (e&7) + ((e>>3)<<4) + ((lane>>4)<<3) relative to the 32-wide k-tile.
// Both chunks are 8 bf16 = 16B; second chunk is +16 elements (= +2 uint4).

__device__ inline bf16x16 frag_ld_lds(const __bf16* base, int ld, int k0, int lane) {
    FragU f;
    const uint4* p = (const uint4*)(base + (lane & 15) * ld + k0 + ((lane >> 4) << 3));
    f.q[0] = p[0];
    f.q[1] = p[2];
    return f.v;
}

__device__ inline bf16x16 frag_ld_glb(const __bf16* __restrict__ W, int ld,
                                      int row0, int k0, int lane) {
    FragU f;
    const uint4* p = (const uint4*)(W + (size_t)(row0 + (lane & 15)) * ld + k0 +
                                    ((lane >> 4) << 3));
    f.q[0] = p[0];
    f.q[1] = p[2];
    return f.v;
}

__device__ inline f32x8 wmma_bf16(bf16x16 a, bf16x16 b, f32x8 c) {
    return __builtin_amdgcn_wmma_f32_16x16x32_bf16(false, a, false, b, (short)0, c,
                                                   false, false);
}

__device__ inline float sigm(float x)      { return 1.0f / (1.0f + __expf(-x)); }
__device__ inline float tanh_fast(float x) { return 2.0f / (1.0f + __expf(-2.0f * x)) - 1.0f; }

// A-fragment for the prenet-1 input, built straight from global f32 with
// zero padding for k>=400 and the t==0 teacher-forcing zero frame.
__device__ inline bf16x16 frag_x(const float* __restrict__ inputs, int b0, int t,
                                 int k0, int lane) {
    bf16x16 r;
    int m  = lane & 15;
    int hi = (lane >> 4) << 3;
    const float* row = inputs + ((size_t)(b0 + m) * TDEC + (t > 0 ? t - 1 : 0)) * INR;
#pragma unroll
    for (int e = 0; e < 16; ++e) {
        int k = k0 + hi + ((e & 7) + ((e >> 3) << 4));
        float v = 0.f;
        if (t > 0 && k < INR) v = row[k];
        r[e] = (__bf16)v;
    }
    return r;
}

// One GRU's 6 gate GEMM accumulations, one 16-col N-tile per wave, fused in regs.
template <int KT_I>
__device__ inline void gru_wmma(const __bf16* Abf, int lda, const __bf16* Hbf,
                                const __bf16* __restrict__ Wih, int ldwi,
                                const __bf16* __restrict__ Whh,
                                const float* __restrict__ bih,
                                const float* __restrict__ bhh,
                                int n0, int nl, int lane,
                                f32x8& sr, f32x8& sz, f32x8& si, f32x8& sh) {
    f32x8 ir, iz, in_, hr, hz, hn;
    float bir = bih[n0 + nl], biz = bih[256 + n0 + nl], bin = bih[512 + n0 + nl];
    float bhr = bhh[n0 + nl], bhz = bhh[256 + n0 + nl], bhn = bhh[512 + n0 + nl];
#pragma unroll
    for (int i = 0; i < 8; ++i) {
        ir[i] = bir; iz[i] = biz; in_[i] = bin;
        hr[i] = bhr; hz[i] = bhz; hn[i] = bhn;
    }
#pragma unroll
    for (int kt = 0; kt < KT_I; ++kt) {
        bf16x16 a = frag_ld_lds(Abf, lda, kt * 32, lane);
        ir  = wmma_bf16(a, frag_ld_glb(Wih, ldwi, n0,       kt * 32, lane), ir);
        iz  = wmma_bf16(a, frag_ld_glb(Wih, ldwi, 256 + n0, kt * 32, lane), iz);
        in_ = wmma_bf16(a, frag_ld_glb(Wih, ldwi, 512 + n0, kt * 32, lane), in_);
    }
#pragma unroll
    for (int kt = 0; kt < 8; ++kt) {
        bf16x16 a = frag_ld_lds(Hbf, 256, kt * 32, lane);
        hr = wmma_bf16(a, frag_ld_glb(Whh, 256, n0,       kt * 32, lane), hr);
        hz = wmma_bf16(a, frag_ld_glb(Whh, 256, 256 + n0, kt * 32, lane), hz);
        hn = wmma_bf16(a, frag_ld_glb(Whh, 256, 512 + n0, kt * 32, lane), hn);
    }
    sr = ir + hr;
    sz = iz + hz;
    si = in_;
    sh = hn;
}

// ---------------------------------------------------------------------------
// Kernel 0: f32 -> bf16 weight conversion with optional K padding.
// ---------------------------------------------------------------------------
__global__ void cvt_pad_kernel(const float* __restrict__ src, __bf16* __restrict__ dst,
                               int rows, int K, int Kp) {
    int total = rows * Kp;
    for (int idx = blockIdx.x * blockDim.x + threadIdx.x; idx < total;
         idx += gridDim.x * blockDim.x) {
        int r = idx / Kp;
        int k = idx - r * Kp;
        float v = (k < K) ? src[(size_t)r * K + k] : 0.f;
        dst[idx] = (__bf16)v;
    }
}

// ---------------------------------------------------------------------------
// Kernel 1: keys = enc @ Wk^T (one big GEMM), also writes bf16 copy of enc.
// Block = 512 threads (16 waves), 64 rows of the 65536 x 256 problem.
// ---------------------------------------------------------------------------
__global__ __launch_bounds__(512)
void keys_kernel(const float* __restrict__ enc, const __bf16* __restrict__ WkBf,
                 __bf16* __restrict__ encBf, float* __restrict__ keys) {
    __shared__ __align__(16) __bf16 sA[64 * 256];
    const int r0   = blockIdx.x * 64;
    const int tid  = threadIdx.x;
    const int lane = tid & 31;
    const int wave = tid >> 5;

    for (int i = tid; i < 64 * 256; i += 512) {
        int r = i >> 8, d = i & 255;
        float  v = enc[(size_t)(r0 + r) * HID + d];
        __bf16 b = (__bf16)v;
        sA[i] = b;
        encBf[(size_t)(r0 + r) * HID + d] = b;
    }
    __syncthreads();

    const int n0 = wave * 16;
    const int nl = lane & 15;
    const int mb = (lane >> 4) << 3;
#pragma unroll 1
    for (int mt = 0; mt < 4; ++mt) {
        f32x8 acc = {};
#pragma unroll
        for (int kt = 0; kt < 8; ++kt)
            acc = wmma_bf16(frag_ld_lds(sA + mt * 16 * 256, 256, kt * 32, lane),
                            frag_ld_glb(WkBf, 256, n0, kt * 32, lane), acc);
#pragma unroll
        for (int i = 0; i < 8; ++i)
            keys[(size_t)(r0 + mt * 16 + mb + i) * HID + n0 + nl] = acc[i];
    }
}

// ---------------------------------------------------------------------------
// Kernel 2: persistent per-batch-tile decoder. Grid = 8 blocks x 512 threads.
// ---------------------------------------------------------------------------
__global__ __launch_bounds__(512)
void decoder_kernel(const float* __restrict__ inputs, const float* __restrict__ keys,
                    const __bf16* __restrict__ encBf, const __bf16* __restrict__ wb,
                    const float* __restrict__ pb1,   const float* __restrict__ pb2,
                    const float* __restrict__ abih,  const float* __restrict__ abhh,
                    const float* __restrict__ bq,    const float* __restrict__ attnv,
                    const float* __restrict__ bout,
                    const float* __restrict__ d1bih, const float* __restrict__ d1bhh,
                    const float* __restrict__ d2bih, const float* __restrict__ d2bhh,
                    const float* __restrict__ projb, float* __restrict__ out) {
    // ~61 KB static LDS (safe even under a 64 KB CU-mode cap).
    __shared__ __align__(16) __bf16 sArena[16 * 256];  // p1 / ctx / d_in (bf16)
    __shared__ __align__(16) __bf16 sCat[16 * 384];    // [p2 | prev attn_out] bf16
    __shared__ __align__(16) __bf16 sHaBf[16 * 256];
    __shared__ __align__(16) __bf16 sH1Bf[16 * 256];
    __shared__ __align__(16) __bf16 sH2Bf[16 * 256];
    __shared__ __align__(16) float  sQ[16 * 256];
    __shared__ float sV[256];

    const __bf16* w1    = wb + OFF_W1;
    const __bf16* w2    = wb + OFF_W2;
    const __bf16* aih   = wb + OFF_AIH;
    const __bf16* ahh   = wb + OFF_AHH;
    const __bf16* wq    = wb + OFF_WQ;
    const __bf16* wout  = wb + OFF_WOUT;
    const __bf16* d1ih  = wb + OFF_D1IH;
    const __bf16* d1hh  = wb + OFF_D1HH;
    const __bf16* d2ih  = wb + OFF_D2IH;
    const __bf16* d2hh  = wb + OFF_D2HH;
    const __bf16* wproj = wb + OFF_PROJ;

    const int b0   = blockIdx.x * 16;
    const int tid  = threadIdx.x;
    const int lane = tid & 31;
    const int wave = tid >> 5;
    const int n0   = wave * 16;
    const int nl   = lane & 15;
    const int n    = n0 + nl;
    const int mb   = (lane >> 4) << 3;

    // init states
    for (int i = tid; i < 16 * 384; i += 512) sCat[i] = (__bf16)0.f;
    for (int i = tid; i < 16 * 256; i += 512) {
        sHaBf[i] = (__bf16)0.f;
        sH1Bf[i] = (__bf16)0.f;
        sH2Bf[i] = (__bf16)0.f;
    }
    for (int i = tid; i < 256; i += 512) sV[i] = attnv[i];
    __syncthreads();

#pragma unroll 1
    for (int t = 0; t < TDEC; ++t) {
        // -------- prenet layer 1: 400 -> 256, relu (A straight from global)
        {
            float bv = pb1[n];
            f32x8 acc;
#pragma unroll
            for (int i = 0; i < 8; ++i) acc[i] = bv;
#pragma unroll
            for (int kt = 0; kt < 13; ++kt)
                acc = wmma_bf16(frag_x(inputs, b0, t, kt * 32, lane),
                                frag_ld_glb(w1, INRP, n0, kt * 32, lane), acc);
#pragma unroll
            for (int i = 0; i < 8; ++i)
                sArena[(mb + i) * 256 + n] = (__bf16)fmaxf(acc[i], 0.f);
        }
        __syncthreads();

        // -------- prenet layer 2: 256 -> 128, relu  (waves 0..7)
        if (wave < 8) {
            float bv = pb2[n];
            f32x8 acc;
#pragma unroll
            for (int i = 0; i < 8; ++i) acc[i] = bv;
#pragma unroll
            for (int kt = 0; kt < 8; ++kt)
                acc = wmma_bf16(frag_ld_lds(sArena, 256, kt * 32, lane),
                                frag_ld_glb(w2, 256, n0, kt * 32, lane), acc);
#pragma unroll
            for (int i = 0; i < 8; ++i)
                sCat[(mb + i) * 384 + n] = (__bf16)fmaxf(acc[i], 0.f);
        }
        __syncthreads();

        // -------- attention GRU: input = [p2 | prev attn_out] (K=384)
        {
            f32x8 sr, sz, si, sh;
            gru_wmma<12>(sCat, 384, sHaBf, aih, 384, ahh, abih, abhh,
                         n0, nl, lane, sr, sz, si, sh);
            __syncthreads();
#pragma unroll
            for (int i = 0; i < 8; ++i) {
                float r  = sigm(sr[i]);
                float z  = sigm(sz[i]);
                float nn = tanh_fast(si[i] + r * sh[i]);
                float ho = (float)sHaBf[(mb + i) * 256 + n];
                sHaBf[(mb + i) * 256 + n] = (__bf16)((1.f - z) * nn + z * ho);
            }
        }
        __syncthreads();

        // -------- q = h_a @ Wq^T + bq
        {
            float bv = bq[n];
            f32x8 acc;
#pragma unroll
            for (int i = 0; i < 8; ++i) acc[i] = bv;
#pragma unroll
            for (int kt = 0; kt < 8; ++kt)
                acc = wmma_bf16(frag_ld_lds(sHaBf, 256, kt * 32, lane),
                                frag_ld_glb(wq, 256, n0, kt * 32, lane), acc);
#pragma unroll
            for (int i = 0; i < 8; ++i) sQ[(mb + i) * 256 + n] = acc[i];
        }
        __syncthreads();

        // -------- scores -> softmax -> context, all wave-private (row = wave)
        {
            const int w = wave;
            float aln[16];
            const float* krow0 = keys + (size_t)(b0 + w) * TENC * HID;
#pragma unroll 1
            for (int c = 0; c < 16; ++c) {
                int te = c * 32 + lane;
                const float* kr = krow0 + (size_t)te * HID;
                float s = 0.f;
#pragma unroll 1
                for (int d = 0; d < HID; d += 8) {
                    float4 ka = *(const float4*)(kr + d);
                    float4 kb = *(const float4*)(kr + d + 4);
                    s += sV[d + 0] * tanh_fast(ka.x + sQ[w * HID + d + 0]);
                    s += sV[d + 1] * tanh_fast(ka.y + sQ[w * HID + d + 1]);
                    s += sV[d + 2] * tanh_fast(ka.z + sQ[w * HID + d + 2]);
                    s += sV[d + 3] * tanh_fast(ka.w + sQ[w * HID + d + 3]);
                    s += sV[d + 4] * tanh_fast(kb.x + sQ[w * HID + d + 4]);
                    s += sV[d + 5] * tanh_fast(kb.y + sQ[w * HID + d + 5]);
                    s += sV[d + 6] * tanh_fast(kb.z + sQ[w * HID + d + 6]);
                    s += sV[d + 7] * tanh_fast(kb.w + sQ[w * HID + d + 7]);
                }
                aln[c] = s;
            }
            // softmax across 512 (16 regs x 32 lanes)
            float mx = -3.0e38f;
#pragma unroll
            for (int c = 0; c < 16; ++c) mx = fmaxf(mx, aln[c]);
            for (int o = 16; o > 0; o >>= 1) mx = fmaxf(mx, __shfl_xor(mx, o, 32));
            float sum = 0.f;
#pragma unroll
            for (int c = 0; c < 16; ++c) { aln[c] = __expf(aln[c] - mx); sum += aln[c]; }
            for (int o = 16; o > 0; o >>= 1) sum += __shfl_xor(sum, o, 32);
            float inv = 1.f / sum;
            size_t abase = ALIGN_OFF + ((size_t)(b0 + w) * TDEC + t) * TENC;
#pragma unroll
            for (int c = 0; c < 16; ++c) {
                aln[c] *= inv;
                out[abase + c * 32 + lane] = aln[c];
            }
            // ctx[w][lane*8 .. +7] = sum_te align * enc_bf
            float acc[8] = {0.f, 0.f, 0.f, 0.f, 0.f, 0.f, 0.f, 0.f};
            const __bf16* eb = encBf + (size_t)(b0 + w) * TENC * HID + lane * 8;
#pragma unroll 1
            for (int te = 0; te < TENC; ++te) {
                float a = __shfl(aln[te >> 5], te & 31, 32);
                Bf8U u;
                u.q = *(const uint4*)(eb + (size_t)te * HID);
#pragma unroll
                for (int cc = 0; cc < 8; ++cc) acc[cc] += a * (float)u.h[cc];
            }
#pragma unroll
            for (int cc = 0; cc < 8; ++cc)
                sArena[wave * 256 + lane * 8 + cc] = (__bf16)acc[cc];
        }
        __syncthreads();

        // -------- attn_out = [ctx | h_a] @ Wout^T + bout (keep f32 in regs)
        f32x8 attnAcc;
        {
            float bv = bout[n];
#pragma unroll
            for (int i = 0; i < 8; ++i) attnAcc[i] = bv;
#pragma unroll
            for (int kt = 0; kt < 8; ++kt)
                attnAcc = wmma_bf16(frag_ld_lds(sArena, 256, kt * 32, lane),
                                    frag_ld_glb(wout, 512, n0, kt * 32, lane), attnAcc);
#pragma unroll
            for (int kt = 0; kt < 8; ++kt)
                attnAcc = wmma_bf16(frag_ld_lds(sHaBf, 256, kt * 32, lane),
                                    frag_ld_glb(wout, 512, n0, 256 + kt * 32, lane), attnAcc);
#pragma unroll
            for (int i = 0; i < 8; ++i)
                sCat[(mb + i) * 384 + 128 + n] = (__bf16)attnAcc[i];  // next prev_ctx + dec1 input
        }
        __syncthreads();

        // -------- decoder GRU 1 (+ residual d_in = h1 + attn_out)
        {
            f32x8 sr, sz, si, sh;
            gru_wmma<8>(sCat + 128, 384, sH1Bf, d1ih, 256, d1hh, d1bih, d1bhh,
                        n0, nl, lane, sr, sz, si, sh);
            __syncthreads();
#pragma unroll
            for (int i = 0; i < 8; ++i) {
                float r  = sigm(sr[i]);
                float z  = sigm(sz[i]);
                float nn = tanh_fast(si[i] + r * sh[i]);
                float ho = (float)sH1Bf[(mb + i) * 256 + n];
                float h1 = (1.f - z) * nn + z * ho;
                sH1Bf[(mb + i) * 256 + n]  = (__bf16)h1;
                sArena[(mb + i) * 256 + n] = (__bf16)(h1 + attnAcc[i]);  // d_in
            }
        }
        __syncthreads();

        // -------- decoder GRU 2
        {
            f32x8 sr, sz, si, sh;
            gru_wmma<8>(sArena, 256, sH2Bf, d2ih, 256, d2hh, d2bih, d2bhh,
                        n0, nl, lane, sr, sz, si, sh);
            __syncthreads();
#pragma unroll
            for (int i = 0; i < 8; ++i) {
                float r  = sigm(sr[i]);
                float z  = sigm(sz[i]);
                float nn = tanh_fast(si[i] + r * sh[i]);
                float ho = (float)sH2Bf[(mb + i) * 256 + n];
                sH2Bf[(mb + i) * 256 + n] = (__bf16)((1.f - z) * nn + z * ho);
            }
        }
        __syncthreads();

        // -------- projection 256 -> 400 (25 N-tiles over 16 waves)
        for (int j = wave; j < 25; j += 16) {
            int jn0 = j * 16;
            float bv = projb[jn0 + nl];
            f32x8 acc;
#pragma unroll
            for (int i = 0; i < 8; ++i) acc[i] = bv;
#pragma unroll
            for (int kt = 0; kt < 8; ++kt)
                acc = wmma_bf16(frag_ld_lds(sH2Bf, 256, kt * 32, lane),
                                frag_ld_glb(wproj, 256, jn0, kt * 32, lane), acc);
#pragma unroll
            for (int i = 0; i < 8; ++i)
                out[((size_t)(b0 + mb + i) * TDEC + t) * INR + jn0 + nl] = acc[i];
        }
        __syncthreads();
    }
}

// ---------------------------------------------------------------------------
extern "C" void kernel_launch(void* const* d_in, const int* in_sizes, int n_in,
                              void* d_out, int out_size, void* d_ws, size_t ws_size,
                              hipStream_t stream) {
    (void)in_sizes; (void)n_in; (void)out_size; (void)ws_size;
    const float* enc    = (const float*)d_in[0];
    const float* inputs = (const float*)d_in[1];

    uint8_t* ws   = (uint8_t*)d_ws;
    float*  keys  = (float*)ws;                              // 64 MB
    __bf16* encBf = (__bf16*)(ws + KEYS_BYTES);              // 32 MB
    __bf16* wb    = (__bf16*)(ws + KEYS_BYTES + ENCBF_BYTES);// ~3.6 MB

    struct CvtJob { int idx; size_t off; int rows, K, Kp; };
    const CvtJob jobs[12] = {
        {2,  OFF_W1,   256, 400, 416},
        {4,  OFF_W2,   128, 256, 256},
        {6,  OFF_AIH,  768, 384, 384},
        {7,  OFF_AHH,  768, 256, 256},
        {10, OFF_WQ,   256, 256, 256},
        {12, OFF_WK,   256, 256, 256},
        {14, OFF_WOUT, 256, 512, 512},
        {16, OFF_D1IH, 768, 256, 256},
        {17, OFF_D1HH, 768, 256, 256},
        {20, OFF_D2IH, 768, 256, 256},
        {21, OFF_D2HH, 768, 256, 256},
        {24, OFF_PROJ, 400, 256, 256},
    };
    for (int j = 0; j < 12; ++j) {
        int total  = jobs[j].rows * jobs[j].Kp;
        int blocks = (total + 255) / 256;
        cvt_pad_kernel<<<blocks, 256, 0, stream>>>(
            (const float*)d_in[jobs[j].idx], wb + jobs[j].off,
            jobs[j].rows, jobs[j].K, jobs[j].Kp);
    }

    keys_kernel<<<(B_TOT * TENC) / 64, 512, 0, stream>>>(enc, wb + OFF_WK, encBf, keys);

    decoder_kernel<<<8, 512, 0, stream>>>(
        inputs, keys, encBf, wb,
        (const float*)d_in[3],  (const float*)d_in[5],
        (const float*)d_in[8],  (const float*)d_in[9],
        (const float*)d_in[11], (const float*)d_in[13],
        (const float*)d_in[15],
        (const float*)d_in[18], (const float*)d_in[19],
        (const float*)d_in[22], (const float*)d_in[23],
        (const float*)d_in[25],
        (float*)d_out);
}
